// MultiHeadAttention_90091234000900
// MI455X (gfx1250) — compile-verified, matrix operand order checked by reading, not run on hardware
//
#include <hip/hip_runtime.h>

#define B_ 2
#define S_ 2048
#define F_ 1024
#define H_ 16
#define D_ 64

typedef __attribute__((ext_vector_type(16))) _Float16 v16h;
typedef __attribute__((ext_vector_type(8)))  _Float16 v8h;
typedef __attribute__((ext_vector_type(8)))  float    v8f;
typedef __attribute__((ext_vector_type(4)))  unsigned v4u;
typedef __attribute__((ext_vector_type(8)))  int      v8i;
typedef __attribute__((ext_vector_type(4)))  int      v4i;

// Load one 16x32 f16 WMMA A/B fragment from a row-major tile (ld = row stride
// in elements). Per-lane layout (CDNA5 ISA 7.12.2): lane holds row (lane&15);
// halves 0..7 = K = kbase..kbase+7, halves 8..15 = K = kbase+16..kbase+23,
// with kbase = (lane>>4)*8.  Two 16B loads per lane.
__device__ __forceinline__ v16h load_frag(const _Float16* tile, int ld, int lane) {
  const int r = lane & 15;
  const int g = lane >> 4;
  const _Float16* p = tile + r * ld + g * 8;
  v8h lo = *(const v8h*)(p);
  v8h hi = *(const v8h*)(p + 16);
  v16h f;
#pragma unroll
  for (int i = 0; i < 8; ++i) { f[i] = lo[i]; f[i + 8] = hi[i]; }
  return f;
}

__device__ __forceinline__ v8f wmma_f16(v16h a, v16h b, v8f c) {
  return __builtin_amdgcn_wmma_f32_16x16x32_f16(false, a, false, b, (short)0, c,
                                                false, false);
}

// ---------------- staging: fp32 -> f16 (and transposed weights) -------------

__global__ void k_cvt_f32_f16(const float* __restrict__ in,
                              _Float16* __restrict__ out, int n) {
  int i = blockIdx.x * blockDim.x + threadIdx.x;
  const int stride = gridDim.x * blockDim.x;
  for (; i < n; i += stride) out[i] = (_Float16)in[i];
}

// in: [K][N] f32   ->   out: [N][K] f16
__global__ void k_transpose_f32_f16(const float* __restrict__ in,
                                    _Float16* __restrict__ out, int K, int N) {
  int i = blockIdx.x * blockDim.x + threadIdx.x;
  const int stride = gridDim.x * blockDim.x;
  const int total = K * N;
  for (; i < total; i += stride) {
    int nn = i / K;
    int kk = i - nn * K;
    out[i] = (_Float16)in[kk * N + nn];
  }
}

// ---------------- QKV GEMM: qkv = x @ w_attn + b_attn -----------------------
// Block = 8 waves sharing one 16-row A slab (staged to LDS by the Tensor Data
// Mover); each wave computes a 16x64 output strip (A fragment reused x4).
// Scatter into per-head Q (scaled 1/sqrt(D)), K, and transposed Vt.

__global__ __launch_bounds__(256) void k_qkv(
    const _Float16* __restrict__ xh, const _Float16* __restrict__ wt,
    const float* __restrict__ bias, _Float16* __restrict__ Q,
    _Float16* __restrict__ K, _Float16* __restrict__ Vt) {
  __shared__ __align__(16) _Float16 Atile[16 * F_];  // 32 KB
  const int lane = threadIdx.x & 31;
  const int wv = threadIdx.x >> 5;
  const int mt = blockIdx.x / 6;             // 256 M-tiles
  const int nblk = blockIdx.x - mt * 6;      // 6 x 512-wide N macro-tiles
  const int m0 = mt * 16;
  const int n0 = nblk * 512 + wv * 64;
  const int r = lane & 15, g = lane >> 4;

  // ---- stage A slab (rows m0..m0+15, all K) into LDS via the TDM ----
#if defined(__has_builtin) && __has_builtin(__builtin_amdgcn_tensor_load_to_lds)
  if (wv == 0) {
    const unsigned lds = (unsigned)(size_t)&Atile[0];  // addr[31:0] = LDS offset
    const unsigned long long ga =
        (unsigned long long)(size_t)(xh + (size_t)m0 * F_);
    v4u g0;
    g0[0] = 1u;                                   // count=1, user descriptor
    g0[1] = lds;                                  // lds_addr
    g0[2] = (unsigned)ga;                         // global_addr[31:0]
    g0[3] = (unsigned)((ga >> 32) & 0x1FFFFFFu) | (2u << 30);  // addr[56:32] | type=2
    v8i g1;
    g1[0] = 0x00010000;                 // data_size = 2 bytes
    g1[1] = (int)(1024u << 16);         // tensor_dim0[15:0] = 1024
    g1[2] = (int)(16u << 16);           // tensor_dim0 hi = 0, tensor_dim1 = 16
    g1[3] = (int)(1024u << 16);         // tensor_dim1 hi = 0, tile_dim0 = 1024
    g1[4] = 16;                         // tile_dim1 = 16, tile_dim2 = 0
    g1[5] = 1024;                       // tensor_dim0_stride[31:0] = 1024
    g1[6] = 0;                          // stride hi, tensor_dim1_stride lo = 0
    g1[7] = 0;
    v4i g2 = {};
    v4i g3 = {};
    v8i g4 = {};
    __builtin_amdgcn_tensor_load_to_lds(g0, g1, g2, g3, g4, 0);
    __builtin_amdgcn_s_wait_tensorcnt(0);
  }
#else
  {  // cooperative fallback: slab is contiguous (full rows)
    const v8h* src = (const v8h*)(xh + (size_t)m0 * F_);
    v8h* dst = (v8h*)&Atile[0];
    for (int idx = threadIdx.x; idx < (16 * F_) / 8; idx += 256)
      dst[idx] = src[idx];
  }
#endif
  __syncthreads();

  v8f acc[4] = {};
  const _Float16* Bb = wt + (size_t)n0 * F_;
#pragma unroll 2
  for (int k0 = 0; k0 < F_; k0 += 32) {
    v16h a = load_frag(&Atile[k0], F_, lane);               // from LDS
    v16h b0 = load_frag(Bb + k0, F_, lane);
    v16h b1 = load_frag(Bb + 16 * F_ + k0, F_, lane);
    v16h b2 = load_frag(Bb + 32 * F_ + k0, F_, lane);
    v16h b3 = load_frag(Bb + 48 * F_ + k0, F_, lane);
    acc[0] = wmma_f16(a, b0, acc[0]);
    acc[1] = wmma_f16(a, b1, acc[1]);
    acc[2] = wmma_f16(a, b2, acc[2]);
    acc[3] = wmma_f16(a, b3, acc[3]);
  }

#pragma unroll
  for (int j = 0; j < 4; ++j) {
    const int e = n0 + j * 16 + r;     // output column in [0, 3F)
    const float bb = bias[e];
    const int sel = e >> 10;           // 0 = Q, 1 = K, 2 = V (uniform per wave)
    const int ee = e & (F_ - 1);
    const int h = ee >> 6;
    const int d = ee & 63;
#pragma unroll
    for (int i = 0; i < 8; ++i) {
      const int mm = m0 + i + g * 8;   // global row in [0, B*S)
      const int b_ = mm >> 11;         // / S_
      const int s = mm & (S_ - 1);
      const int bh = b_ * H_ + h;
      const float v = acc[j][i] + bb;
      if (sel == 0)
        Q[((size_t)bh * S_ + s) * D_ + d] = (_Float16)(v * 0.125f);  // 1/sqrt(64)
      else if (sel == 1)
        K[((size_t)bh * S_ + s) * D_ + d] = (_Float16)v;
      else
        Vt[((size_t)bh * D_ + d) * S_ + s] = (_Float16)v;
    }
  }
}

// ---------------- causal flash attention ------------------------------------
// One wave per (b, h, 16-row q tile).  Online softmax, P staged via LDS to
// convert C-layout -> A-fragment layout.

__global__ __launch_bounds__(256) void k_attn(const _Float16* __restrict__ Q,
                                              const _Float16* __restrict__ K,
                                              const _Float16* __restrict__ Vt,
                                              _Float16* __restrict__ Zm) {
  __shared__ __align__(16) _Float16 P[8][16 * 32];
  const int lane = threadIdx.x & 31;
  const int wv = threadIdx.x >> 5;
  const int tile = blockIdx.x * 8 + wv;
  const int qt = tile & (S_ / 16 - 1);   // 128 q-tiles
  const int bh = tile >> 7;              // [0, B*H)
  const int b = bh >> 4, h = bh & 15;
  const int q0 = qt * 16;
  const int r = lane & 15, g = lane >> 4;

  const _Float16* Qb = Q + ((size_t)bh * S_ + q0) * D_;
  const v16h qf0 = load_frag(Qb, D_, lane);        // K-dim (D) chunk 0..31
  const v16h qf1 = load_frag(Qb + 32, D_, lane);   // chunk 32..63

  v8f z0 = {}, z1 = {}, z2 = {}, z3 = {};
  float mr[8], lr[8];
#pragma unroll
  for (int i = 0; i < 8; ++i) { mr[i] = -1e30f; lr[i] = 0.f; }

  const int nsteps = (qt + 2) >> 1;      // causal: keys up to q0+15
  for (int st = 0; st < nsteps; ++st) {
    const int kkb = st * 32;
    const _Float16* Kb = K + ((size_t)bh * S_ + kkb) * D_;
    v16h kf00 = load_frag(Kb, D_, lane);
    v16h kf01 = load_frag(Kb + 32, D_, lane);
    v16h kf10 = load_frag(Kb + 16 * D_, D_, lane);
    v16h kf11 = load_frag(Kb + 16 * D_ + 32, D_, lane);
    v8f s0 = {}, s1 = {};
    s0 = wmma_f16(qf0, kf00, s0);
    s0 = wmma_f16(qf1, kf01, s0);
    s1 = wmma_f16(qf0, kf10, s1);
    s1 = wmma_f16(qf1, kf11, s1);

    const int kk0 = kkb + r, kk1 = kkb + 16 + r;
#pragma unroll
    for (int i = 0; i < 8; ++i) {
      const int qg = q0 + g * 8 + i;
      float v0 = (kk0 <= qg) ? s0[i] : -1e30f;
      float v1 = (kk1 <= qg) ? s1[i] : -1e30f;
      float mx = fmaxf(v0, v1);
      mx = fmaxf(mx, __shfl_xor(mx, 1, 32));  // reduce within 16-lane row group
      mx = fmaxf(mx, __shfl_xor(mx, 2, 32));
      mx = fmaxf(mx, __shfl_xor(mx, 4, 32));
      mx = fmaxf(mx, __shfl_xor(mx, 8, 32));
      const float mnew = fmaxf(mr[i], mx);
      const float corr = __expf(mr[i] - mnew);
      const float p0 = __expf(v0 - mnew);
      const float p1 = __expf(v1 - mnew);
      float rs = p0 + p1;
      rs += __shfl_xor(rs, 1, 32);
      rs += __shfl_xor(rs, 2, 32);
      rs += __shfl_xor(rs, 4, 32);
      rs += __shfl_xor(rs, 8, 32);
      lr[i] = lr[i] * corr + rs;
      mr[i] = mnew;
      z0[i] *= corr; z1[i] *= corr; z2[i] *= corr; z3[i] *= corr;
      P[wv][(i + g * 8) * 32 + r] = (_Float16)p0;        // row-major 16x32
      P[wv][(i + g * 8) * 32 + 16 + r] = (_Float16)p1;
    }
    // in-wave DS store -> DS load fence (DS ops are in-order per wave; this
    // also keeps the compiler from hoisting the fragment loads).
    asm volatile("s_wait_dscnt 0" ::: "memory");
    v16h pf = load_frag(&P[wv][0], 32, lane);

    const _Float16* Vb = Vt + (size_t)bh * D_ * S_ + kkb;  // Vt[d][s]
    v16h vf0 = load_frag(Vb, S_, lane);
    v16h vf1 = load_frag(Vb + 16 * S_, S_, lane);
    v16h vf2 = load_frag(Vb + 32 * S_, S_, lane);
    v16h vf3 = load_frag(Vb + 48 * S_, S_, lane);
    z0 = wmma_f16(pf, vf0, z0);
    z1 = wmma_f16(pf, vf1, z1);
    z2 = wmma_f16(pf, vf2, z2);
    z3 = wmma_f16(pf, vf3, z3);
  }

#pragma unroll
  for (int i = 0; i < 8; ++i) {
    const float inv = 1.0f / lr[i];
    const int q = q0 + i + g * 8;
    _Float16* o = Zm + ((size_t)(b * S_ + q)) * F_ + h * 64 + r;
    o[0]  = (_Float16)(z0[i] * inv);
    o[16] = (_Float16)(z1[i] * inv);
    o[32] = (_Float16)(z2[i] * inv);
    o[48] = (_Float16)(z3[i] * inv);
  }
}

// ---------------- output projection: out = merged @ w_proj + b_proj ---------
// Each wave computes a 16x64 strip (A fragment reused x4).

__global__ __launch_bounds__(256) void k_proj(const _Float16* __restrict__ Zm,
                                              const _Float16* __restrict__ wpt,
                                              const float* __restrict__ bias,
                                              float* __restrict__ out) {
  const int lane = threadIdx.x & 31;
  const int wv = threadIdx.x >> 5;
  const int tile = blockIdx.x * 8 + wv;  // 4096 strips
  const int mt = tile >> 4;              // 256 M-tiles
  const int nw = tile & 15;              // 16 x 64-wide N strips
  const int m0 = mt * 16, n0 = nw * 64;
  const int r = lane & 15, g = lane >> 4;

  v8f acc[4] = {};
  const _Float16* Ab = Zm + (size_t)m0 * F_;
  const _Float16* Bb = wpt + (size_t)n0 * F_;
#pragma unroll 2
  for (int k0 = 0; k0 < F_; k0 += 32) {
    v16h a = load_frag(Ab + k0, F_, lane);
    v16h b0 = load_frag(Bb + k0, F_, lane);
    v16h b1 = load_frag(Bb + 16 * F_ + k0, F_, lane);
    v16h b2 = load_frag(Bb + 32 * F_ + k0, F_, lane);
    v16h b3 = load_frag(Bb + 48 * F_ + k0, F_, lane);
    acc[0] = wmma_f16(a, b0, acc[0]);
    acc[1] = wmma_f16(a, b1, acc[1]);
    acc[2] = wmma_f16(a, b2, acc[2]);
    acc[3] = wmma_f16(a, b3, acc[3]);
  }
#pragma unroll
  for (int j = 0; j < 4; ++j) {
    const float bb = bias[n0 + j * 16 + r];
#pragma unroll
    for (int i = 0; i < 8; ++i) {
      const int mm = m0 + i + g * 8;
      out[(size_t)mm * F_ + n0 + j * 16 + r] = acc[j][i] + bb;
    }
  }
}

// ---------------- host launch ----------------------------------------------

extern "C" void kernel_launch(void* const* d_in, const int* in_sizes, int n_in,
                              void* d_out, int out_size, void* d_ws,
                              size_t ws_size, hipStream_t stream) {
  (void)in_sizes; (void)n_in; (void)out_size; (void)ws_size;
  const float* x      = (const float*)d_in[0];
  const float* w_attn = (const float*)d_in[1];
  const float* b_attn = (const float*)d_in[2];
  const float* w_proj = (const float*)d_in[3];
  const float* b_proj = (const float*)d_in[4];
  float* out = (float*)d_out;

  char* ws = (char*)d_ws;
  size_t off = 0;
  auto take = [&](size_t bytes) -> char* {
    char* p = ws + off;
    off += (bytes + 255) & ~(size_t)255;
    return p;
  };
  _Float16* xh  = (_Float16*)take((size_t)B_ * S_ * F_ * 2);        // 8 MB
  _Float16* wat = (_Float16*)take((size_t)3 * F_ * F_ * 2);         // 6 MB
  _Float16* wpt = (_Float16*)take((size_t)F_ * F_ * 2);             // 2 MB
  _Float16* Q   = (_Float16*)take((size_t)B_ * H_ * S_ * D_ * 2);   // 8 MB
  _Float16* K   = (_Float16*)take((size_t)B_ * H_ * S_ * D_ * 2);   // 8 MB
  _Float16* Vt  = (_Float16*)take((size_t)B_ * H_ * S_ * D_ * 2);   // 8 MB
  _Float16* Zm  = (_Float16*)take((size_t)B_ * S_ * F_ * 2);        // 8 MB

  k_cvt_f32_f16<<<2048, 256, 0, stream>>>(x, xh, B_ * S_ * F_);
  k_transpose_f32_f16<<<2048, 256, 0, stream>>>(w_attn, wat, F_, 3 * F_);
  k_transpose_f32_f16<<<1024, 256, 0, stream>>>(w_proj, wpt, F_, F_);
  // 256 M-tiles x 6 N macro-tiles, 8 waves/block (each wave: 16x64 strip)
  k_qkv<<<1536, 256, 0, stream>>>(xh, wat, b_attn, Q, K, Vt);
  // B*H*128 = 4096 q-tiles / 8 waves per block
  k_attn<<<512, 256, 0, stream>>>(Q, K, Vt, Zm);
  // 256 M-tiles x 16 N strips / 8 waves per block
  k_proj<<<512, 256, 0, stream>>>(Zm, wpt, b_proj, out);
}
